// GraphormerExplainer_25812753449667
// MI455X (gfx1250) — matrix operand order repaired and checked by reading, not compiled, not run on hardware
//
#include <hip/hip_runtime.h>
#include <hip/hip_bf16.h>

typedef __attribute__((ext_vector_type(16))) __bf16 v16bf;
typedef __attribute__((ext_vector_type(4)))  __bf16 v4bf;
typedef __attribute__((ext_vector_type(8)))  float  v8f;
typedef __attribute__((ext_vector_type(4)))  unsigned int uint4v;

union V16 {
    uint4v q[2];
    v16bf  v;
};

#define PAD 136   // 128 + 8 bf16 row padding (272 B rows -> conflict-free column access)

// One 16x128 @ 128x128 layer on WMMA bf16, f32 accumulate.
// aBase: this lane's feat/h row in LDS (row M = lane&15, padded stride PAD)
// wt:    transposed weights in LDS, [n][k], row stride PAD
// bs[8]: per-lane bias splat values (bias[nt*16 + (lane&15)])
__device__ __forceinline__ void gemm_layer(const __bf16* aBase, const __bf16* wt,
                                           const float bs[8], v8f acc[8],
                                           int ln, int hi) {
    V16 a[4];
#pragma unroll
    for (int kc = 0; kc < 4; ++kc) {
        // 16-bit A 16x32 layout: lanes 0-15 K = {0..7,16..23}, lanes 16-31 K = {8..15,24..31}
        const __bf16* p = aBase + kc * 32 + hi * 8;
        a[kc].q[0] = *(const uint4v*)p;
        a[kc].q[1] = *(const uint4v*)(p + 16);
    }
#pragma unroll
    for (int nt = 0; nt < 8; ++nt) {
        v8f c;
#pragma unroll
        for (int j = 0; j < 8; ++j) c[j] = bs[nt];
#pragma unroll
        for (int kc = 0; kc < 4; ++kc) {
            // 16-bit B 32x16 layout: lane = column N, lanes 0-15 K = {0..7,16..23}, lanes 16-31 K = {8..15,24..31}
            const __bf16* bp = wt + (nt * 16 + ln) * PAD + kc * 32 + hi * 8;
            V16 b;
            b.q[0] = *(const uint4v*)bp;
            b.q[1] = *(const uint4v*)(bp + 16);
            c = __builtin_amdgcn_wmma_f32_16x16x32_bf16(
                    false, a[kc].v, false, b.v, (short)0, c, false, false);
        }
        acc[nt] = c;
    }
}

// relu + f32->bf16, scatter D-layout (M = r + 8*hi, N = nt*16 + ln) back to row-major LDS.
__device__ __forceinline__ void relu_store(const v8f acc[8], __bf16* mb, int ln, int hi) {
#pragma unroll
    for (int nt = 0; nt < 8; ++nt)
#pragma unroll
        for (int r = 0; r < 8; ++r) {
            float v = acc[nt][r];
            v = v > 0.f ? v : 0.f;
            mb[(r + hi * 8) * PAD + nt * 16 + ln] = (__bf16)v;
        }
}

__global__ __launch_bounds__(256)
void edge_mlp_kernel(const float* __restrict__ node,
                     const long long* __restrict__ ei,   // [2, E] int64
                     const float* __restrict__ W1, const float* __restrict__ b1,
                     const float* __restrict__ W2, const float* __restrict__ b2,
                     const float* __restrict__ W3, const float* __restrict__ b3,
                     float* __restrict__ out, int E, int nTiles) {
    __shared__ __bf16 w1t[128 * PAD];          // W1^T bf16, [n][k]
    __shared__ __bf16 w2t[128 * PAD];          // W2^T bf16
    __shared__ __bf16 buf[8 * 16 * PAD];       // per-wave 16x128 staging (feat, then h1)

    // Stage transposed bf16 weights once per block.
    for (int i = threadIdx.x; i < 128 * 128; i += 256) {
        int k = i >> 7, n = i & 127;           // W row-major [k][n]
        w1t[n * PAD + k] = (__bf16)W1[i];
        w2t[n * PAD + k] = (__bf16)W2[i];
    }
    __syncthreads();

    const int lane = threadIdx.x & 31;
    const int wave = threadIdx.x >> 5;
    const int ln   = lane & 15;
    const int hi   = lane >> 4;
    __bf16* mb = buf + wave * 16 * PAD;

    float b1s[8], b2s[8], w3s[8];
#pragma unroll
    for (int nt = 0; nt < 8; ++nt) {
        b1s[nt] = b1[nt * 16 + ln];
        b2s[nt] = b2[nt * 16 + ln];
        w3s[nt] = W3[nt * 16 + ln];
    }
    const float bias3 = b3[0];

    for (int t = blockIdx.x; t < nTiles; t += gridDim.x) {
        const int e0 = t * 128 + wave * 16;

        // Lanes 0-15 hold this tile's edge indices (node ids < 100000 fit in int).
        int sidx = 0, didx = 0;
        if (lane < 16) {
            int e = e0 + lane;
            if (e >= E) e = E - 1;
            sidx = (int)ei[e];
            didx = (int)ei[E + e];
        }

        // Gather: one feat row (src 64 | dst 64 floats) per iteration, all 32 lanes.
#pragma unroll 4
        for (int r = 0; r < 16; ++r) {
            int s = __shfl(sidx, r);
            int d = __shfl(didx, r);
            const float* rowp = (lane < 16)
                ? (node + (size_t)s * 64 + lane * 4)
                : (node + (size_t)d * 64 + (lane - 16) * 4);
            float4 v = *(const float4*)rowp;
            v4bf o = { (__bf16)v.x, (__bf16)v.y, (__bf16)v.z, (__bf16)v.w };
            *(v4bf*)(mb + r * PAD + lane * 4) = o;
        }

        v8f acc[8];
        // Layer 1: relu(feat @ W1 + b1)
        gemm_layer(mb + ln * PAD, w1t, b1s, acc, ln, hi);
        relu_store(acc, mb, ln, hi);
        // Layer 2: relu(h1 @ W2 + b2)
        gemm_layer(mb + ln * PAD, w2t, b2s, acc, ln, hi);

        // Layer 3: relu(h2) @ W3 + b3, sigmoid, store.
#pragma unroll
        for (int r = 0; r < 8; ++r) {
            float p = 0.f;
#pragma unroll
            for (int nt = 0; nt < 8; ++nt) {
                float v = acc[nt][r];
                v = v > 0.f ? v : 0.f;          // relu after layer 2
                p += v * w3s[nt];
            }
            // reduce over N within each 16-lane half
            p += __shfl_xor(p, 1);
            p += __shfl_xor(p, 2);
            p += __shfl_xor(p, 4);
            p += __shfl_xor(p, 8);
            if (ln == r) {
                int e = e0 + hi * 8 + r;        // M = r + 8*hi
                if (e < E) {
                    float x = p + bias3;
                    out[e] = 1.0f / (1.0f + __expf(-x));
                }
            }
        }
    }
}

extern "C" void kernel_launch(void* const* d_in, const int* in_sizes, int n_in,
                              void* d_out, int out_size, void* d_ws, size_t ws_size,
                              hipStream_t stream) {
    (void)n_in; (void)d_ws; (void)ws_size; (void)out_size;
    const float*     node = (const float*)d_in[0];
    const long long* ei   = (const long long*)d_in[1];
    const float*     W1   = (const float*)d_in[2];
    const float*     b1   = (const float*)d_in[3];
    const float*     W2   = (const float*)d_in[4];
    const float*     b2   = (const float*)d_in[5];
    const float*     W3   = (const float*)d_in[6];
    const float*     b3   = (const float*)d_in[7];
    float* out = (float*)d_out;

    const int E = in_sizes[1] / 2;             // edge_index is [2, E]
    const int nTiles = (E + 127) / 128;        // 128 edges per block iteration
    int blocks = nTiles < 1024 ? nTiles : 1024;

    edge_mlp_kernel<<<blocks, 256, 0, stream>>>(node, ei, W1, b1, W2, b2, W3, b3,
                                                out, E, nTiles);
}
